// CrossAttention_10101763080198
// MI455X (gfx1250) — compile-verified, matrix operand order checked by reading, not run on hardware
//
#include <hip/hip_runtime.h>

// ---------------------------------------------------------------------------
// Flash-attention forward (B=2, Sq=2048, Sk=2048, H=16, D=64), fp32 I/O.
// CDNA5 / gfx1250, wave32, bf16 WMMA + f32 accumulation.
//
//  Kernel 1 (pre-pass): kv f32 -> bf16 once. K row-major [b][h][s][d],
//    V transposed [b][h][d][s], mask -> additive f32 (log2 domain).
//  Kernel 2 (main): per 64-key block, stage K/V/mask straight into LDS with
//    GLOBAL_LOAD_ASYNC_TO_LDS (ASYNCcnt DMA, no VGPR staging), double
//    buffered; 18 WMMAs per block (8 QK + 8 PV + 2 row-sum-vs-ones);
//    row max via ds_swizzle butterfly; P relayout via packed-dword LDS.
// ---------------------------------------------------------------------------

typedef __attribute__((ext_vector_type(16))) __bf16   v16bf;
typedef __attribute__((ext_vector_type(8)))  __bf16   v8bf;
typedef __attribute__((ext_vector_type(2)))  __bf16   v2bf;
typedef __attribute__((ext_vector_type(8)))  float    v8f;
typedef __attribute__((ext_vector_type(8)))  unsigned int v8u;

constexpr int B_  = 2;
constexpr int SQ  = 2048;
constexpr int SK  = 2048;
constexpr int H_  = 16;
constexpr int D_  = 64;

constexpr int QTILE = 16;              // q rows per wave
constexpr int WAVES = 8;               // 256 threads = 8 wave32
constexpr int QBLK  = QTILE * WAVES;   // 128 q rows per block
constexpr int KBLK  = 64;              // keys per inner iteration
constexpr int NKB   = SK / KBLK;       // 32

// LDS strides (bf16 elems): multiples of 8 -> 16B alignment for b128 ops;
// row stride 144B = 36 banks -> conflict-free 16-lane tile reads.
constexpr int KS_STRIDE = D_  + 8;     // 72
constexpr int VT_STRIDE = KBLK + 8;    // 72
constexpr int PS_STRIDE = KBLK + 8;    // 72

#define LOG2E   1.4426950408889634f
#define NEG_BIG (-10000.0f * LOG2E)    // additive mask, log2 domain

// ---------------- CDNA5 helpers ----------------

static __device__ __forceinline__ v8f wmma_bf16(v16bf a, v16bf b, v8f c) {
  return __builtin_amdgcn_wmma_f32_16x16x32_bf16(
      false, a, false, b, (short)0, c, false, false);
}

static __device__ __forceinline__ v16bf cat8(v8bf lo, v8bf hi) {
  v16bf r;
  #pragma unroll
  for (int i = 0; i < 8; ++i) { r[i] = lo[i]; r[8 + i] = hi[i]; }
  return r;
}

// max all-reduce over the 16 lanes of each half (ds_swizzle SWAPX1/2/4/8)
static __device__ __forceinline__ float halfmax16(float t) {
  t = fmaxf(t, __int_as_float(__builtin_amdgcn_ds_swizzle(__float_as_int(t), 0x041F)));
  t = fmaxf(t, __int_as_float(__builtin_amdgcn_ds_swizzle(__float_as_int(t), 0x081F)));
  t = fmaxf(t, __int_as_float(__builtin_amdgcn_ds_swizzle(__float_as_int(t), 0x101F)));
  t = fmaxf(t, __int_as_float(__builtin_amdgcn_ds_swizzle(__float_as_int(t), 0x201F)));
  return t;
}

// GLOBAL_LOAD_ASYNC_TO_LDS_*: per-lane async DMA global -> LDS (ASYNCcnt).
// VDST = LDS byte address (low 32 bits of the flat shared pointer).
static __device__ __forceinline__ void async_b128(const void* g, void* l) {
  unsigned int       loff = (unsigned int)(unsigned long long)l;
  unsigned long long ga   = (unsigned long long)g;
  asm volatile("global_load_async_to_lds_b128 %0, %1, off"
               :: "v"(loff), "v"(ga) : "memory");
}
static __device__ __forceinline__ void async_b32(const void* g, void* l) {
  unsigned int       loff = (unsigned int)(unsigned long long)l;
  unsigned long long ga   = (unsigned long long)g;
  asm volatile("global_load_async_to_lds_b32 %0, %1, off"
               :: "v"(loff), "v"(ga) : "memory");
}
static __device__ __forceinline__ void wait_async0() {
  asm volatile("s_wait_asynccnt 0" ::: "memory");
}

// ---------------- pre-pass: kv f32 -> bf16 (K row-major, V transposed) -----

__global__ __launch_bounds__(256, 1)
void prep_kv(const float* __restrict__ kv, const unsigned char* __restrict__ kpm,
             __bf16* __restrict__ Kbf, __bf16* __restrict__ Vtb,
             float* __restrict__ Mf)
{
  __shared__ __align__(16) __bf16 T[KBLK][VT_STRIDE];   // V transpose tile

  const int tid = threadIdx.x;
  const int bid = blockIdx.x;
  const int skb = bid & (SK / KBLK - 1);     // 0..31
  const int hd  = (bid >> 5) & (H_ - 1);
  const int b   = bid >> 9;

  const int row = tid >> 2;                  // s within block (also d on write)
  const int cg  = (tid & 3) * 16;
  const int s   = skb * KBLK + row;

  const float* kp = kv + ((((size_t)b * SK + s) * 2 + 0) * H_ + hd) * D_ + cg;
  const float* vp = kv + ((((size_t)b * SK + s) * 2 + 1) * H_ + hd) * D_ + cg;

  // K: straight convert, coalesced both sides
  v8bf k0, k1;
  #pragma unroll
  for (int i = 0; i < 8; ++i) { k0[i] = (__bf16)kp[i]; k1[i] = (__bf16)kp[8 + i]; }
  __bf16* kd = Kbf + (((size_t)b * H_ + hd) * SK + s) * D_ + cg;
  *(v8bf*)kd       = k0;
  *(v8bf*)(kd + 8) = k1;

  // V: transpose 64x64 through LDS, write [d][s] coalesced
  #pragma unroll
  for (int i = 0; i < 8; ++i) {
    T[cg + i][row]     = (__bf16)vp[i];
    T[cg + 8 + i][row] = (__bf16)vp[8 + i];
  }
  __syncthreads();
  v8bf t0 = *(const v8bf*)&T[row][cg];
  v8bf t1 = *(const v8bf*)&T[row][cg + 8];
  __bf16* vd = Vtb + (((size_t)b * H_ + hd) * D_ + row) * SK + skb * KBLK + cg;
  *(v8bf*)vd       = t0;
  *(v8bf*)(vd + 8) = t1;

  // mask -> additive float in log2 domain (once per (b, s))
  if (hd == 0 && tid < KBLK) {
    const size_t ms = (size_t)b * SK + skb * KBLK + tid;
    Mf[ms] = kpm[ms] ? 0.0f : NEG_BIG;
  }
}

// ---------------- main flash-attention kernel ----------------

__global__ __launch_bounds__(256, 1)
void fa_fwd_bf16(const float* __restrict__ q,
                 const __bf16* __restrict__ Kbf,
                 const __bf16* __restrict__ Vtb,
                 const float* __restrict__ Mf,
                 float* __restrict__ out)
{
  __shared__ __align__(16) __bf16 Ks[2][KBLK][KS_STRIDE];       // K, row-major
  __shared__ __align__(16) __bf16 Vt[2][D_][VT_STRIDE];         // V, transposed
  __shared__ __align__(16) __bf16 Ps[WAVES][QTILE][PS_STRIDE];  // packed P dwords
  __shared__ __align__(16) float  Ms[2][KBLK];                  // additive mask

  const int tid  = threadIdx.x;
  const int w    = tid >> 5;
  const int lane = tid & 31;
  const int n    = lane & 15;          // tile column
  const int hf   = lane >> 4;          // WMMA half selector

  const int bid = blockIdx.x;
  const int qb  = bid & 15;            // SQ/QBLK == 16
  const int hd  = (bid >> 4) & (H_ - 1);
  const int b   = bid >> 8;
  const int bh  = b * H_ + hd;

  const int q0 = qb * QBLK + w * QTILE;
  const float qscale = LOG2E / 8.0f;   // log2(e)/sqrt(D)

  // ---- Q tile in WMMA A layout: two 16x32 bf16 chunks over D=64 ----
  v16bf Qa[2];
  {
    const float* qrow = q + (((size_t)b * SQ + q0 + n) * H_ + hd) * D_;
    #pragma unroll
    for (int c = 0; c < 2; ++c) {
      const int d0 = 32 * c;
      #pragma unroll
      for (int i = 0; i < 8; ++i) {
        Qa[c][i]     = (__bf16)(qrow[d0 + 8 * hf + i]      * qscale);
        Qa[c][8 + i] = (__bf16)(qrow[d0 + 16 + 8 * hf + i] * qscale);
      }
    }
  }

  // all-ones B tile: rowsum(P) = P @ ones via WMMA
  v16bf onesB;
  #pragma unroll
  for (int i = 0; i < 16; ++i) onesB[i] = (__bf16)1.0f;

  v8f O[4] = {};
  float m_r[8], l_r[8];
  #pragma unroll
  for (int r = 0; r < 8; ++r) { m_r[r] = -3.0e38f; l_r[r] = 0.0f; }

  // ---- async staging: global bf16 -> LDS, no VGPR trip ----
  auto stage_async = [&](int kb, int ib) {
    const int row = tid >> 2;          // key row (K) / d row (Vt)
    const int cg  = (tid & 3) * 16;
    const __bf16* ksrc = Kbf + (((size_t)bh) * SK + kb * KBLK + row) * D_ + cg;
    const __bf16* vsrc = Vtb + (((size_t)bh) * D_ + row) * SK + kb * KBLK + cg;
    async_b128(ksrc,     &Ks[ib][row][cg]);
    async_b128(ksrc + 8, &Ks[ib][row][cg + 8]);
    async_b128(vsrc,     &Vt[ib][row][cg]);
    async_b128(vsrc + 8, &Vt[ib][row][cg + 8]);
    if (kb + 1 < NKB) {                // warm L2 one block further ahead
      __builtin_prefetch(ksrc + (size_t)KBLK * D_, 0, 3);
      __builtin_prefetch(vsrc + KBLK, 0, 3);
    }
    if (tid < KBLK)
      async_b32(Mf + (size_t)b * SK + kb * KBLK + tid, &Ms[ib][tid]);
  };

  // ---- per-wave compute on one staged 64-key block ----
  auto compute = [&](int ib) {
    // S = Q K^T : four 16x16 f32 tiles
    v8f S[4];
    #pragma unroll
    for (int c = 0; c < 4; ++c) {
      const __bf16* kr = &Ks[ib][c * 16 + n][0];
      v8f acc = wmma_bf16(Qa[0],
                          cat8(*(const v8bf*)&kr[8 * hf],
                               *(const v8bf*)&kr[16 + 8 * hf]),
                          (v8f){});
      acc = wmma_bf16(Qa[1],
                      cat8(*(const v8bf*)&kr[32 + 8 * hf],
                           *(const v8bf*)&kr[48 + 8 * hf]),
                      acc);
      const float madd = Ms[ib][c * 16 + n];
      #pragma unroll
      for (int r = 0; r < 8; ++r) acc[r] += madd;
      S[c] = acc;
    }

    // online softmax: row max via ds_swizzle butterfly (rows = r + 8*hf)
    float alpha[8];
    #pragma unroll
    for (int r = 0; r < 8; ++r) {
      float t = fmaxf(fmaxf(S[0][r], S[1][r]), fmaxf(S[2][r], S[3][r]));
      t = halfmax16(t);
      const float mnew = fmaxf(m_r[r], t);
      alpha[r] = exp2f(m_r[r] - mnew);
      m_r[r]   = mnew;
      #pragma unroll
      for (int c = 0; c < 4; ++c) S[c][r] = exp2f(S[c][r] - mnew);
    }
    #pragma unroll
    for (int t = 0; t < 4; ++t)
      #pragma unroll
      for (int r = 0; r < 8; ++r) O[t][r] *= alpha[r];

    // relayout P: store column pair (col, col+16) packed in one dword ->
    // 16 b32 stores; reload 2x b128 per chunk + v_perm deinterleave.
    #pragma unroll
    for (int ch = 0; ch < 2; ++ch)
      #pragma unroll
      for (int r = 0; r < 8; ++r) {
        v2bf p;
        p[0] = (__bf16)S[2 * ch][r];
        p[1] = (__bf16)S[2 * ch + 1][r];
        *(v2bf*)&Ps[w][r + 8 * hf][ch * 32 + 2 * n] = p;
      }
    asm volatile("s_wait_dscnt 0" ::: "memory");   // cross-lane LDS RAW in-wave

    v16bf Pa[2];
    #pragma unroll
    for (int ch = 0; ch < 2; ++ch) {
      const uint4 lo = *(const uint4*)&Ps[w][n][ch * 32 + 16 * hf];
      const uint4 hi = *(const uint4*)&Ps[w][n][ch * 32 + 16 * hf + 8];
      const unsigned int d[8] = {lo.x, lo.y, lo.z, lo.w, hi.x, hi.y, hi.z, hi.w};
      v8u pu;
      #pragma unroll
      for (int j = 0; j < 4; ++j) {
        pu[j]     = __builtin_amdgcn_perm(d[2 * j + 1], d[2 * j], 0x05040100u);
        pu[4 + j] = __builtin_amdgcn_perm(d[2 * j + 1], d[2 * j], 0x07060302u);
      }
      Pa[ch] = __builtin_bit_cast(v16bf, pu);
    }

    // row sums: P @ ones (2 WMMAs instead of a shuffle butterfly)
    v8f Rv = wmma_bf16(Pa[0], onesB, (v8f){});
    Rv     = wmma_bf16(Pa[1], onesB, Rv);
    #pragma unroll
    for (int r = 0; r < 8; ++r) l_r[r] = l_r[r] * alpha[r] + Rv[r];

    // O += P V  (V pre-transposed -> contiguous B tiles)
    #pragma unroll
    for (int t = 0; t < 4; ++t) {
      const __bf16* vr = &Vt[ib][t * 16 + n][0];
      #pragma unroll
      for (int ch = 0; ch < 2; ++ch) {
        O[t] = wmma_bf16(Pa[ch],
                         cat8(*(const v8bf*)&vr[ch * 32 + 8 * hf],
                              *(const v8bf*)&vr[ch * 32 + 16 + 8 * hf]),
                         O[t]);
      }
    }
  };

  // ---- main loop: async staging double-buffered against compute ----
  stage_async(0, 0);
  wait_async0();
  __syncthreads();

  for (int kb = 0; kb < NKB; ++kb) {
    if (kb + 1 < NKB) stage_async(kb + 1, (kb + 1) & 1);  // DMA during compute
    compute(kb & 1);
    wait_async0();                     // our async writes landed in LDS
    __syncthreads();                   // everyone's writes visible
  }

  // ---- normalize, write fp32 output (b, t, h, d) ----
  #pragma unroll
  for (int r = 0; r < 8; ++r) {
    const float inv = 1.0f / l_r[r];
    float* orow = out + (((size_t)b * SQ + q0 + r + 8 * hf) * H_ + hd) * D_;
    #pragma unroll
    for (int t = 0; t < 4; ++t) orow[t * 16 + n] = O[t][r] * inv;
  }
}

// ---------------- host launcher ----------------

extern "C" void kernel_launch(void* const* d_in, const int* in_sizes, int n_in,
                              void* d_out, int out_size, void* d_ws, size_t ws_size,
                              hipStream_t stream) {
  const float* q          = (const float*)d_in[0];
  const float* kv         = (const float*)d_in[1];
  const unsigned char* km = (const unsigned char*)d_in[2];  // jnp.bool_ -> 1 byte
  float* out              = (float*)d_out;

  // workspace layout: Kbf | Vtb | Mf  (~16.8 MB, L2-resident)
  const size_t kvelems = (size_t)B_ * H_ * SK * D_;
  __bf16* Kbf = (__bf16*)d_ws;
  __bf16* Vtb = Kbf + kvelems;
  float*  Mf  = (float*)(Vtb + kvelems);

  prep_kv<<<dim3(B_ * H_ * (SK / KBLK)), dim3(256), 0, stream>>>(kv, km, Kbf, Vtb, Mf);
  fa_fwd_bf16<<<dim3(B_ * H_ * (SQ / QBLK)), dim3(256), 0, stream>>>(q, Kbf, Vtb, Mf, out);
}
